// STBlock_84980222918803
// MI455X (gfx1250) — matrix operand order repaired and checked by reading, not compiled, not run on hardware
//
#include <hip/hip_runtime.h>
#include <math.h>

// ---------------- problem constants ----------------
constexpr int cB = 16, cN = 325, cC = 64, cT = 32;
constexpr int cV = cB * cN;          // 5200 nodes
constexpr int cR = cV * cT;          // 166400 rows (v*T + t)
constexpr int cE1 = 83200;           // GCN edges
constexpr int cM = 1024;             // hyperedges
constexpr int cNNZ = 41600;          // incidences
constexpr int cTC = cT * cC;         // 2048 feats per node

// ---------------- WMMA types / helpers ----------------
typedef __attribute__((ext_vector_type(16))) _Float16 v16h;
typedef __attribute__((ext_vector_type(8)))  _Float16 v8h;
typedef __attribute__((ext_vector_type(4)))  _Float16 v4h;
typedef __attribute__((ext_vector_type(8)))  float    v8f;

union AF { v16h v; _Float16 h[16]; };

__device__ inline v8f wmma16(v16h a, v16h b, v8f c) {
  // D = A(16x32,f16) * B(32x16,f16) + C(16x16,f32)
  return __builtin_amdgcn_wmma_f32_16x16x32_f16(false, a, false, b, (short)0, c, false, false);
}

// A fragment (16x32 f16). lane row = lane&15; k = (lane>>4)*16 + v*2 + j (matches packed B).
// => per lane the 16 halves are two contiguous 8-half (16B) runs: kbase+half*16 and kbase+half*16+8.
__device__ inline v16h load_a16(const _Float16* rowbase, int stride, int kbase) {
  const int lane = threadIdx.x & 31;
  const _Float16* rp = rowbase + (size_t)(lane & 15) * stride + kbase + ((lane >> 4) << 4);
  v8h lo = *(const v8h*)rp;
  v8h hi = *(const v8h*)(rp + 8);
  return __builtin_shufflevector(lo, hi, 0, 1, 2, 3, 4, 5, 6, 7, 8, 9, 10, 11, 12, 13, 14, 15);
}

// Packed B fragments: [frag][lane][16 halves] contiguous -> one 32B v16h load per lane.
__device__ inline v16h load_b_packed(const _Float16* packed, int fid) {
  return *(const v16h*)(packed + (size_t)fid * 512 + (size_t)(threadIdx.x & 31) * 16);
}

// ---- B fragment builders (used only by the tiny pack kernels) ----
// B[k][n], lane: n = lane&15, k = (lane>>4)*16 + v*2 + j.
__device__ inline v16h build_b_oc(const float* W, int ldo, int nbase, int kbase) {
  const int lane = threadIdx.x & 31;
  const float* wp = W + (size_t)(nbase + (lane & 15)) * ldo + kbase + ((lane >> 4) << 4);
  AF f;
#pragma unroll
  for (int v = 0; v < 16; ++v) f.h[v] = (_Float16)wp[v];
  return f.v;
}
__device__ inline v16h build_b_co(const float* W, int ldn, int nbase, int kbase) {
  const int lane = threadIdx.x & 31;
  const int n = nbase + (lane & 15);
  const int kh = kbase + ((lane >> 4) << 4);
  AF f;
#pragma unroll
  for (int v = 0; v < 16; ++v) f.h[v] = (_Float16)W[(size_t)(kh + v) * ldn + n];
  return f.v;
}
__device__ inline v16h build_b_conv(const float* Wc, int cin, int kt, int nbase, int kbase) {
  const int lane = threadIdx.x & 31;
  const float* wp = Wc + (size_t)(nbase + (lane & 15)) * cin * 3
                       + (size_t)(kbase + ((lane >> 4) << 4)) * 3 + kt;
  AF f;
#pragma unroll
  for (int v = 0; v < 16; ++v) f.h[v] = (_Float16)wp[v * 3];
  return f.v;
}

// ---------------- weight fragment pack kernels (1 wave per fragment) ----------------
__global__ void pack_b_oc_kernel(const float* __restrict__ W, int ldo, int ksteps,
                                 _Float16* __restrict__ out) {
  const int fid = blockIdx.x;
  const int nt = fid / ksteps, ks = fid - nt * ksteps;
  v16h b = build_b_oc(W, ldo, nt * 16, ks * 32);
  *(v16h*)(out + (size_t)fid * 512 + (size_t)(threadIdx.x & 31) * 16) = b;
}
__global__ void pack_b_co_kernel(const float* __restrict__ W, int ldn, int ksteps,
                                 _Float16* __restrict__ out) {
  const int fid = blockIdx.x;
  const int nt = fid / ksteps, ks = fid - nt * ksteps;
  v16h b = build_b_co(W, ldn, nt * 16, ks * 32);
  *(v16h*)(out + (size_t)fid * 512 + (size_t)(threadIdx.x & 31) * 16) = b;
}
__global__ void pack_b_conv_kernel(const float* __restrict__ Wc, int cin, int ksteps,
                                   _Float16* __restrict__ out) {
  const int fid = blockIdx.x;                 // ((ot*3 + kt)*ksteps + ks)
  const int tmp = fid / ksteps, ks = fid - tmp * ksteps;
  const int ot = tmp / 3, kt = tmp - ot * 3;
  v16h b = build_b_conv(Wc, cin, kt, ot * 16, ks * 32);
  *(v16h*)(out + (size_t)fid * 512 + (size_t)(threadIdx.x & 31) * 16) = b;
}

// ---------------- layout conversion: x [B,N,C,T] f32 -> [V,T,C] f16 ----------------
__global__ void cvt_x_kernel(const float* __restrict__ x, _Float16* __restrict__ x16) {
  size_t idx = (size_t)blockIdx.x * 256 + threadIdx.x;   // over cR*16 (4 ch per thread)
  if (idx >= (size_t)cR * 16) return;
  const int c4 = (int)(idx & 15) * 4;
  const size_t r = idx >> 4;          // v*T + t
  const int t = (int)(r & 31);
  const size_t v = r >> 5;
  const float* xp = x + v * cTC + (size_t)c4 * cT + t;
  v4h h = {(_Float16)xp[0], (_Float16)xp[32], (_Float16)xp[64], (_Float16)xp[96]};
  *(v4h*)(x16 + r * 64 + c4) = h;
}

// ---------------- GTU temporal conv (WMMA). One node per block, 4 waves. ----------------
template <int CIN, bool FINAL>
__global__ __launch_bounds__(128) void tconv_wmma_kernel(
    const _Float16* __restrict__ act,                      // [V, T, CIN] f16
    const _Float16* __restrict__ pWa, const float* __restrict__ ba,
    const _Float16* __restrict__ pWc, const float* __restrict__ bc,
    _Float16* __restrict__ out16,
    const float* __restrict__ xres,
    const float* __restrict__ lnw, const float* __restrict__ lnb,
    float* __restrict__ outf) {
  constexpr int SP = CIN + 8;                 // multiple of 8 halves -> 16B-aligned rows
  constexpr int KS = CIN / 32;
  __shared__ __align__(32) _Float16 s_act[34 * SP];   // rows t=-2..31 (causal left pad)
  __shared__ float s_h[32 * 65];
  const int v = blockIdx.x;
  const int tid = threadIdx.x;

  for (int idx = tid; idx < 34 * CIN; idx += 128) {
    const int row = idx / CIN, c = idx - row * CIN;
    const int t = row - 2;
    s_act[row * SP + c] =
        (t >= 0) ? act[(size_t)v * cT * CIN + (size_t)t * CIN + c] : (_Float16)0.f;
  }
  __syncthreads();

  const int wid = tid >> 5, lane = tid & 31;
  const int nlo = lane & 15, hsel = lane >> 4;

  for (int task = wid; task < 8; task += 4) {       // 2 M-tiles x 4 out-col tiles
    const int mtile = task >> 2, otile = task & 3;
    v8f a_xin = {0, 0, 0, 0, 0, 0, 0, 0};
    v8f a_g = a_xin, a_y2 = a_xin;
#pragma unroll
    for (int ks = 0; ks < KS; ++ks) {               // pointwise path (no time shift)
      v16h A = load_a16(s_act + (mtile * 16 + 2) * SP, SP, ks * 32);
      a_xin = wmma16(A, load_b_packed(pWa, otile * KS + ks), a_xin);
    }
#pragma unroll
    for (int kt = 0; kt < 3; ++kt) {                // causal conv: input row t+kt-2
#pragma unroll
      for (int ks = 0; ks < KS; ++ks) {
        v16h A = load_a16(s_act + (mtile * 16 + kt) * SP, SP, ks * 32);
        a_g  = wmma16(A, load_b_packed(pWc, (otile * 3 + kt) * KS + ks), a_g);
        a_y2 = wmma16(A, load_b_packed(pWc, ((otile + 4) * 3 + kt) * KS + ks), a_y2);
      }
    }
    const int o = otile * 16 + nlo;
    const float bax = ba[o], bg0 = bc[o], bg1 = bc[o + 64];
#pragma unroll
    for (int i = 0; i < 8; ++i) {
      const int t = mtile * 16 + hsel * 8 + i;      // C/D layout: m = half*8 + vgpr
      const float xin = a_xin[i] + bax;
      const float g = 1.f / (1.f + __expf(-(a_g[i] + bg0)));
      const float val = (1.f - g) * xin + g * (a_y2[i] + bg1);
      if (FINAL) {
        const float z = val + xres[(size_t)v * cTC + (size_t)o * cT + t];
        s_h[t * 65 + o] = 0.5f * z * (1.f + erff(z * 0.70710678f));   // exact GELU
      } else {
        out16[(size_t)v * cTC + (size_t)t * cC + o] = (_Float16)val;
      }
    }
  }

  if (FINAL) {
    __syncthreads();
    if (tid < 32) {
      const int t = tid;
      float mu = 0.f;
      for (int c = 0; c < 64; ++c) mu += s_h[t * 65 + c];
      mu *= (1.f / 64.f);
      float var = 0.f;
      for (int c = 0; c < 64; ++c) { const float d = s_h[t * 65 + c] - mu; var += d * d; }
      const float inv = rsqrtf(var * (1.f / 64.f) + 1e-5f);
      for (int c = 0; c < 64; ++c)
        outf[(size_t)v * cTC + (size_t)c * cT + t] =
            (s_h[t * 65 + c] - mu) * inv * lnw[c] + lnb[c];
    }
  }
}

// ---------------- [R,64]f16 @ packed(64x64) -> [R,64]f32, 2 M-tiles per wave ----------
__global__ __launch_bounds__(128) void gemm64_wmma_kernel(
    const _Float16* __restrict__ A, const _Float16* __restrict__ pW, float* __restrict__ out) {
  const int wave = blockIdx.x * 4 + (threadIdx.x >> 5);
  const size_t r0 = (size_t)wave * 32;
  const int lane = threadIdx.x & 31;
  v8f z = {0, 0, 0, 0, 0, 0, 0, 0};
  v8f acc0[4] = {z, z, z, z};
  v8f acc1[4] = {z, z, z, z};
#pragma unroll
  for (int ks = 0; ks < 2; ++ks) {
    v16h a0 = load_a16(A + r0 * 64, 64, ks * 32);
    v16h a1 = load_a16(A + (r0 + 16) * 64, 64, ks * 32);
#pragma unroll
    for (int nt = 0; nt < 4; ++nt) {
      v16h b = load_b_packed(pW, nt * 2 + ks);
      acc0[nt] = wmma16(a0, b, acc0[nt]);
      acc1[nt] = wmma16(a1, b, acc1[nt]);
    }
  }
  const int nlo = lane & 15, hsel = lane >> 4;
#pragma unroll
  for (int nt = 0; nt < 4; ++nt)
#pragma unroll
    for (int i = 0; i < 8; ++i) {
      out[(r0 + hsel * 8 + i) * 64 + nt * 16 + nlo] = acc0[nt][i];
      out[(r0 + 16 + hsel * 8 + i) * 64 + nt * 16 + nlo] = acc1[nt][i];
    }
}

// ---------------- CSR build: count -> exclusive scan -> cursor fill ----------------
__global__ void count_kernel(const int* __restrict__ key, int* __restrict__ counts, int n) {
  const int i = blockIdx.x * 256 + threadIdx.x;
  if (i < n) atomicAdd(&counts[key[i]], 1);
}
// Single-block exclusive scan, n <= 5248.
__global__ __launch_bounds__(1024) void scan_excl_kernel(const int* __restrict__ counts,
                                                         int* __restrict__ rowptr, int n) {
  __shared__ int s[5248];
  __shared__ int part[1024];
  const int tid = threadIdx.x;
  const int chunk = (n + 1023) >> 10;
  for (int i = tid; i < n; i += 1024) s[i] = counts[i];
  __syncthreads();
  int loc = 0;
  const int beg = tid * chunk;
  for (int j = 0; j < chunk; ++j) {
    const int i = beg + j;
    if (i < n) loc += s[i];
  }
  part[tid] = loc;
  __syncthreads();
  for (int st = 1; st < 1024; st <<= 1) {           // Hillis-Steele inclusive
    const int val = (tid >= st) ? part[tid - st] : 0;
    __syncthreads();
    part[tid] += val;
    __syncthreads();
  }
  int run = (tid == 0) ? 0 : part[tid - 1];
  for (int j = 0; j < chunk; ++j) {
    const int i = beg + j;
    if (i < n) { const int c = s[i]; rowptr[i] = run; run += c; }
  }
  if (tid == 1023) rowptr[n] = run;
}
__global__ void cpy_int_kernel(int* __restrict__ dst, const int* __restrict__ src, int n) {
  const int i = blockIdx.x * 256 + threadIdx.x;
  if (i < n) dst[i] = src[i];
}
__global__ void fill_kernel(const int* __restrict__ key, int* __restrict__ cursor,
                            int* __restrict__ list, int n) {
  const int i = blockIdx.x * 256 + threadIdx.x;
  if (i < n) list[atomicAdd(&cursor[key[i]], 1)] = i;
}

// ---------------- graph degree / norm ----------------
__global__ void gcn_deg_kernel(const int* __restrict__ dst, const float* __restrict__ w,
                               float* __restrict__ deg, int n) {
  const int i = blockIdx.x * 256 + threadIdx.x;
  if (i < n) atomicAdd(&deg[dst[i]], w[i]);
}
__global__ void gcn_dinv_kernel(float* deg, int n) {  // in-place deg -> dinv (self loop +1)
  const int i = blockIdx.x * 256 + threadIdx.x;
  if (i < n) { const float d = deg[i] + 1.f; deg[i] = (d > 0.f) ? rsqrtf(d) : 0.f; }
}
__global__ void hg_deg_kernel(const int* __restrict__ nidx, const int* __restrict__ eidx,
                              const float* __restrict__ hew, float* __restrict__ Dh,
                              float* __restrict__ Bdeg, int n) {
  const int i = blockIdx.x * 256 + threadIdx.x;
  if (i < n) {
    atomicAdd(&Dh[nidx[i]], hew[eidx[i]]);
    atomicAdd(&Bdeg[eidx[i]], 1.f);
  }
}
__global__ void hg_inv_kernel(float* Dh, float* Bdeg) {  // in-place -> Dinv, Binv
  const int i = blockIdx.x * 256 + threadIdx.x;
  if (i < cV) Dh[i] = (Dh[i] > 0.f) ? 1.f / Dh[i] : 0.f;
  if (i < cM) Bdeg[i] = (Bdeg[i] > 0.f) ? 1.f / Bdeg[i] : 0.f;
}

// ---------------- atomic-free CSR gathers (register accumulation) ----------------
// GCN: out[v,:] = relu( sum_{e: dst=v} dinv[v]*w*dinv[src] * xw[src,:] + dinv[v]^2*xw[v,:] + bg )
__global__ __launch_bounds__(256) void gcn_gather_kernel(
    const int* __restrict__ rowptr, const int* __restrict__ elist,
    const int* __restrict__ esrc, const float* __restrict__ ew,
    const float* __restrict__ dinv, const float* __restrict__ xw,
    const float* __restrict__ bg, float* __restrict__ out) {
  const int v = blockIdx.x;
  const int beg = rowptr[v], end = rowptr[v + 1];
  const float di = dinv[v];
  const int c0 = threadIdx.x * 8;
  const float* xs = xw + (size_t)v * cTC + c0;
  const float sf = di * di;                       // self loop (weight 1)
  float4 a0 = *(const float4*)xs;
  float4 a1 = *(const float4*)(xs + 4);
  a0.x *= sf; a0.y *= sf; a0.z *= sf; a0.w *= sf;
  a1.x *= sf; a1.y *= sf; a1.z *= sf; a1.w *= sf;
  for (int ee = beg; ee < end; ++ee) {
    const int e = elist[ee];
    const int s = esrc[e];
    if (ee + 1 < end)
      __builtin_prefetch(xw + (size_t)esrc[elist[ee + 1]] * cTC + c0, 0, 1);
    const float nrm = di * ew[e] * dinv[s];
    const float* xr = xw + (size_t)s * cTC + c0;
    const float4 b0 = *(const float4*)xr;
    const float4 b1 = *(const float4*)(xr + 4);
    a0.x += nrm * b0.x; a0.y += nrm * b0.y; a0.z += nrm * b0.z; a0.w += nrm * b0.w;
    a1.x += nrm * b1.x; a1.y += nrm * b1.y; a1.z += nrm * b1.z; a1.w += nrm * b1.w;
  }
  float* op = out + (size_t)v * cTC + c0;
  float4 r0, r1;
  r0.x = fmaxf(a0.x + bg[c0 + 0], 0.f); r0.y = fmaxf(a0.y + bg[c0 + 1], 0.f);
  r0.z = fmaxf(a0.z + bg[c0 + 2], 0.f); r0.w = fmaxf(a0.w + bg[c0 + 3], 0.f);
  r1.x = fmaxf(a1.x + bg[c0 + 4], 0.f); r1.y = fmaxf(a1.y + bg[c0 + 5], 0.f);
  r1.z = fmaxf(a1.z + bg[c0 + 6], 0.f); r1.w = fmaxf(a1.w + bg[c0 + 7], 0.f);
  *(float4*)op = r0;
  *(float4*)(op + 4) = r1;
}
// Hypergraph node->hyperedge: m[e,:] = Binv[e] * sum_{i: eidx=e} xw[nidx[i],:]
__global__ __launch_bounds__(256) void hg_edge_kernel(
    const int* __restrict__ rowptr, const int* __restrict__ ilist,
    const int* __restrict__ nidx, const float* __restrict__ Binv,
    const float* __restrict__ xw, float* __restrict__ m) {
  const int e = blockIdx.x;
  const int beg = rowptr[e], end = rowptr[e + 1];
  const float cf = Binv[e];
  const int c0 = threadIdx.x * 8;
  float4 a0 = {0.f, 0.f, 0.f, 0.f}, a1 = {0.f, 0.f, 0.f, 0.f};
  for (int k = beg; k < end; ++k) {
    const int nv = nidx[ilist[k]];
    if (k + 1 < end)
      __builtin_prefetch(xw + (size_t)nidx[ilist[k + 1]] * cTC + c0, 0, 1);
    const float* xr = xw + (size_t)nv * cTC + c0;
    const float4 b0 = *(const float4*)xr;
    const float4 b1 = *(const float4*)(xr + 4);
    a0.x += b0.x; a0.y += b0.y; a0.z += b0.z; a0.w += b0.w;
    a1.x += b1.x; a1.y += b1.y; a1.z += b1.z; a1.w += b1.w;
  }
  a0.x *= cf; a0.y *= cf; a0.z *= cf; a0.w *= cf;
  a1.x *= cf; a1.y *= cf; a1.z *= cf; a1.w *= cf;
  float* op = m + (size_t)e * cTC + c0;
  *(float4*)op = a0;
  *(float4*)(op + 4) = a1;
}
// Hypergraph hyperedge->node: out[v,:] = relu( Dinv[v] * sum_{i: nidx=v} hew[eidx[i]]*m[eidx[i],:] + bh )
__global__ __launch_bounds__(256) void hg_node_kernel(
    const int* __restrict__ rowptr, const int* __restrict__ ilist,
    const int* __restrict__ eidx, const float* __restrict__ hew,
    const float* __restrict__ Dinv, const float* __restrict__ m,
    const float* __restrict__ bh, float* __restrict__ out) {
  const int v = blockIdx.x;
  const int beg = rowptr[v], end = rowptr[v + 1];
  const float di = Dinv[v];
  const int c0 = threadIdx.x * 8;
  float4 a0 = {0.f, 0.f, 0.f, 0.f}, a1 = {0.f, 0.f, 0.f, 0.f};
  for (int k = beg; k < end; ++k) {
    const int e = eidx[ilist[k]];
    if (k + 1 < end)
      __builtin_prefetch(m + (size_t)eidx[ilist[k + 1]] * cTC + c0, 0, 1);
    const float cf = hew[e];
    const float* xr = m + (size_t)e * cTC + c0;
    const float4 b0 = *(const float4*)xr;
    const float4 b1 = *(const float4*)(xr + 4);
    a0.x += cf * b0.x; a0.y += cf * b0.y; a0.z += cf * b0.z; a0.w += cf * b0.w;
    a1.x += cf * b1.x; a1.y += cf * b1.y; a1.z += cf * b1.z; a1.w += cf * b1.w;
  }
  float* op = out + (size_t)v * cTC + c0;
  float4 r0, r1;
  r0.x = fmaxf(a0.x * di + bh[c0 + 0], 0.f); r0.y = fmaxf(a0.y * di + bh[c0 + 1], 0.f);
  r0.z = fmaxf(a0.z * di + bh[c0 + 2], 0.f); r0.w = fmaxf(a0.w * di + bh[c0 + 3], 0.f);
  r1.x = fmaxf(a1.x * di + bh[c0 + 4], 0.f); r1.y = fmaxf(a1.y * di + bh[c0 + 5], 0.f);
  r1.z = fmaxf(a1.z * di + bh[c0 + 6], 0.f); r1.w = fmaxf(a1.w * di + bh[c0 + 7], 0.f);
  *(float4*)op = r0;
  *(float4*)(op + 4) = r1;
}

// ---------------- SE gating ----------------
__global__ __launch_bounds__(256) void se_mean_kernel(
    const float* __restrict__ sx1, const float* __restrict__ sx2, float* __restrict__ mean) {
  const int b = blockIdx.x >> 7, ch = blockIdx.x & 127;
  const float* src = (ch < 64) ? sx1 : sx2;
  const int c = ch & 63;
  const size_t base = (size_t)b * cN * cT * cC + c;
  float s = 0.f;
  for (int r = threadIdx.x; r < cN * cT; r += 256) s += src[base + (size_t)r * 64];
  __shared__ float red[256];
  red[threadIdx.x] = s;
  __syncthreads();
  for (int st = 128; st > 0; st >>= 1) {
    if (threadIdx.x < st) red[threadIdx.x] += red[threadIdx.x + st];
    __syncthreads();
  }
  if (threadIdx.x == 0) mean[blockIdx.x] = red[0] * (1.f / (cN * cT));
}
__global__ __launch_bounds__(128) void se_mlp_kernel(
    const float* __restrict__ mean, const float* __restrict__ Wse1,
    const float* __restrict__ Wse2, float* __restrict__ se) {
  const int b = blockIdx.x, tid = threadIdx.x;
  __shared__ float mv[128];
  __shared__ float hid[8];
  mv[tid] = mean[b * 128 + tid];
  __syncthreads();
  if (tid < 8) {
    float a = 0.f;
    for (int c = 0; c < 128; ++c) a += mv[c] * Wse1[c * 8 + tid];
    hid[tid] = fmaxf(a, 0.f);
  }
  __syncthreads();
  float o = 0.f;
  for (int h = 0; h < 8; ++h) o += hid[h] * Wse2[h * 128 + tid];
  se[b * 128 + tid] = 1.f / (1.f + __expf(-o));
}
__global__ void build_pre16_kernel(const float* __restrict__ sx1, const float* __restrict__ sx2,
                                   const float* __restrict__ se, _Float16* __restrict__ pre) {
  size_t idx = (size_t)blockIdx.x * 256 + threadIdx.x;   // over cR*32 (4 ch per thread)
  if (idx >= (size_t)cR * 32) return;
  const int ch = (int)(idx & 31) * 4;
  const size_t r = idx >> 5;                             // v*T + t
  const int b = (int)(r >> 5) / cN;
  const float4 vv = (ch < 64) ? *(const float4*)(sx1 + r * 64 + ch)
                              : *(const float4*)(sx2 + r * 64 + (ch - 64));
  const float* sep = se + b * 128 + ch;
  v4h h = {(_Float16)(vv.x * sep[0]), (_Float16)(vv.y * sep[1]),
           (_Float16)(vv.z * sep[2]), (_Float16)(vv.w * sep[3])};
  *(v4h*)(pre + r * 128 + ch) = h;
}

// ---------------- launch ----------------
extern "C" void kernel_launch(void* const* d_in, const int* in_sizes, int n_in,
                              void* d_out, int out_size, void* d_ws, size_t ws_size,
                              hipStream_t stream) {
  (void)in_sizes; (void)n_in; (void)out_size; (void)ws_size;
  const float* x    = (const float*)d_in[0];
  const int*   he1  = (const int*)d_in[1];
  const float* hew1 = (const float*)d_in[2];
  const int*   he2  = (const int*)d_in[3];
  const float* hew2 = (const float*)d_in[4];
  const float* Wa1  = (const float*)d_in[5];
  const float* ba1  = (const float*)d_in[6];
  const float* Wc1  = (const float*)d_in[7];
  const float* bc1  = (const float*)d_in[8];
  const float* Wg   = (const float*)d_in[9];
  const float* bg   = (const float*)d_in[10];
  const float* Wh   = (const float*)d_in[11];
  const float* bh   = (const float*)d_in[12];
  const float* Wse1 = (const float*)d_in[13];
  const float* Wse2 = (const float*)d_in[14];
  const float* Wa2  = (const float*)d_in[15];
  const float* ba2  = (const float*)d_in[16];
  const float* Wc2  = (const float*)d_in[17];
  const float* bc2  = (const float*)d_in[18];
  const float* lnw  = (const float*)d_in[19];
  const float* lnb  = (const float*)d_in[20];
  float* outp = (float*)d_out;

  const int* src  = he1;              // HE1[0]
  const int* dst  = he1 + cE1;        // HE1[1]
  const int* nidx = he2;              // HE2[0]
  const int* eidx = he2 + cNNZ;       // HE2[1]

  char* ws = (char*)d_ws;
  size_t off = 0;
  auto take = [&](size_t bytes) -> char* {
    char* p = ws + off;
    off += (bytes + 255) & ~(size_t)255;
    return p;
  };
  _Float16* x16  = (_Float16*)take((size_t)cR * 64 * 2);   // input in [V,T,C] f16
  _Float16* xv16 = (_Float16*)take((size_t)cR * 64 * 2);   // tconv1 out, graph input
  float* xw   = (float*)take((size_t)cR * 64 * 4);         // xv@W (reused as pre16)
  float* sx1  = (float*)take((size_t)cR * 64 * 4);         // GCN output
  float* mbuf = (float*)take((size_t)cM * cTC * 4);        // hyperedge features
  float* sx2  = (float*)take((size_t)cR * 64 * 4);         // HGCN output
  float* deg  = (float*)take(cV * 4);
  float* Dh   = (float*)take(cV * 4);
  float* Bdeg = (float*)take(cM * 4);
  float* semn = (float*)take(2048 * 4);
  float* seve = (float*)take(2048 * 4);
  // packed f16 weight fragments (1KB each)
  _Float16* pWg  = (_Float16*)take(8 * 1024);
  _Float16* pWh  = (_Float16*)take(8 * 1024);
  _Float16* pWa1 = (_Float16*)take(8 * 1024);
  _Float16* pWc1 = (_Float16*)take(48 * 1024);
  _Float16* pWa2 = (_Float16*)take(16 * 1024);
  _Float16* pWc2 = (_Float16*)take(96 * 1024);
  // CSR structures
  int* gcnt = (int*)take(cV * 4);
  int* grp  = (int*)take((cV + 1) * 4);
  int* gcur = (int*)take(cV * 4);
  int* gel  = (int*)take((size_t)cE1 * 4);
  int* hecnt = (int*)take(cM * 4);
  int* herp  = (int*)take((cM + 1) * 4);
  int* hecur = (int*)take(cM * 4);
  int* hel   = (int*)take((size_t)cNNZ * 4);
  int* hncnt = (int*)take(cV * 4);
  int* hnrp  = (int*)take((cV + 1) * 4);
  int* hncur = (int*)take(cV * 4);
  int* hnl   = (int*)take((size_t)cNNZ * 4);
  _Float16* pre16 = (_Float16*)xw;                         // same byte count: R*128*2

  hipMemsetAsync(deg, 0, cV * 4, stream);
  hipMemsetAsync(Dh, 0, cV * 4, stream);
  hipMemsetAsync(Bdeg, 0, cM * 4, stream);
  hipMemsetAsync(gcnt, 0, cV * 4, stream);
  hipMemsetAsync(hecnt, 0, cM * 4, stream);
  hipMemsetAsync(hncnt, 0, cV * 4, stream);

  // weight fragment packing (one wave per 16x16x32 B-fragment)
  pack_b_co_kernel<<<8, 32, 0, stream>>>(Wg, 64, 2, pWg);
  pack_b_co_kernel<<<8, 32, 0, stream>>>(Wh, 64, 2, pWh);
  pack_b_oc_kernel<<<8, 32, 0, stream>>>(Wa1, 64, 2, pWa1);
  pack_b_conv_kernel<<<48, 32, 0, stream>>>(Wc1, 64, 2, pWc1);
  pack_b_oc_kernel<<<16, 32, 0, stream>>>(Wa2, 128, 4, pWa2);
  pack_b_conv_kernel<<<96, 32, 0, stream>>>(Wc2, 128, 4, pWc2);

  // CSR builds (GCN by dst; hypergraph by eidx and by nidx)
  count_kernel<<<(cE1 + 255) / 256, 256, 0, stream>>>(dst, gcnt, cE1);
  scan_excl_kernel<<<1, 1024, 0, stream>>>(gcnt, grp, cV);
  cpy_int_kernel<<<(cV + 255) / 256, 256, 0, stream>>>(gcur, grp, cV);
  fill_kernel<<<(cE1 + 255) / 256, 256, 0, stream>>>(dst, gcur, gel, cE1);
  count_kernel<<<(cNNZ + 255) / 256, 256, 0, stream>>>(eidx, hecnt, cNNZ);
  scan_excl_kernel<<<1, 1024, 0, stream>>>(hecnt, herp, cM);
  cpy_int_kernel<<<(cM + 255) / 256, 256, 0, stream>>>(hecur, herp, cM);
  fill_kernel<<<(cNNZ + 255) / 256, 256, 0, stream>>>(eidx, hecur, hel, cNNZ);
  count_kernel<<<(cNNZ + 255) / 256, 256, 0, stream>>>(nidx, hncnt, cNNZ);
  scan_excl_kernel<<<1, 1024, 0, stream>>>(hncnt, hnrp, cV);
  cpy_int_kernel<<<(cV + 255) / 256, 256, 0, stream>>>(hncur, hnrp, cV);
  fill_kernel<<<(cNNZ + 255) / 256, 256, 0, stream>>>(nidx, hncur, hnl, cNNZ);

  cvt_x_kernel<<<(cR * 16 + 255) / 256, 256, 0, stream>>>(x, x16);

  // tconv1 (GTU) -> xv16 [V,T,C] f16
  tconv_wmma_kernel<64, false><<<cV, 128, 0, stream>>>(
      x16, pWa1, ba1, pWc1, bc1, xv16, nullptr, nullptr, nullptr, nullptr);

  // GCN: norm, xv@Wg (WMMA), atomic-free CSR gather (+ self loop + bias + relu)
  gcn_deg_kernel<<<(cE1 + 255) / 256, 256, 0, stream>>>(dst, hew1, deg, cE1);
  gcn_dinv_kernel<<<(cV + 255) / 256, 256, 0, stream>>>(deg, cV);
  gemm64_wmma_kernel<<<cR / 128, 128, 0, stream>>>(xv16, pWg, xw);
  gcn_gather_kernel<<<cV, 256, 0, stream>>>(grp, gel, src, hew1, deg, xw, bg, sx1);

  // HGCN: degrees, xv@Wh (WMMA), node->hyperedge, hyperedge->node (+ Dinv + bias + relu)
  hg_deg_kernel<<<(cNNZ + 255) / 256, 256, 0, stream>>>(nidx, eidx, hew2, Dh, Bdeg, cNNZ);
  hg_inv_kernel<<<(cV + 255) / 256, 256, 0, stream>>>(Dh, Bdeg);
  gemm64_wmma_kernel<<<cR / 128, 128, 0, stream>>>(xv16, pWh, xw);
  hg_edge_kernel<<<cM, 256, 0, stream>>>(herp, hel, nidx, Bdeg, xw, mbuf);
  hg_node_kernel<<<cV, 256, 0, stream>>>(hnrp, hnl, eidx, hew2, Dh, mbuf, bh, sx2);

  // SE gating + pre (f16, SE-scaled concat)
  se_mean_kernel<<<cB * 128, 256, 0, stream>>>(sx1, sx2, semn);
  se_mlp_kernel<<<cB, 128, 0, stream>>>(semn, Wse1, Wse2, seve);
  build_pre16_kernel<<<(cR * 32 + 255) / 256, 256, 0, stream>>>(sx1, sx2, seve, pre16);

  // tconv2 (GTU) + residual + GELU + LayerNorm -> d_out [B,N,C,T] f32
  tconv_wmma_kernel<128, true><<<cV, 128, 0, stream>>>(
      pre16, pWa2, ba2, pWc2, bc2, nullptr, x, lnw, lnb, outp);
}
